// SelfAttention_81982335746208
// MI455X (gfx1250) — compile-verified, matrix operand order checked by reading, not run on hardware
//
#include <hip/hip_runtime.h>
#include <hip/hip_bf16.h>
#include <stdint.h>

// ---------------------------------------------------------------------------
// SAGAN self-attention, flash-fused for MI455X (gfx1250, wave32, WMMA bf16)
// B=8, C=256, N=4096, R=32.  All WMMA fragments are K-contiguous -> each
// fragment = two b128 loads per lane (no scalar u16 gathers).
// ---------------------------------------------------------------------------

typedef __attribute__((ext_vector_type(16))) __bf16 v16bf;
typedef __attribute__((ext_vector_type(8)))  float  v8f;

constexpr int Bc = 8, Cc = 256, Nc = 4096, Rc = 32;

#define DEV static __device__ __forceinline__

DEV unsigned short f2bf(float f) {          // round-to-nearest-even f32 -> bf16
  unsigned int u = __float_as_uint(f);
  u += 0x7FFFu + ((u >> 16) & 1u);
  return (unsigned short)(u >> 16);
}

union FragU { v16bf v; unsigned int u[8]; uint4 q[2]; };

DEV v8f v8zero() {
  v8f z;
#pragma unroll
  for (int i = 0; i < 8; ++i) z[i] = 0.f;
  return z;
}

// Fragment from a matrix stored [row][K] (row = M for A, row = N for B^T
// storage), K elements contiguous.  16-bit layout (ISA 7.12.2): lane m=L&15,
// h=L>>4; VGPR v holds K = (v<4?0:16) + 8h + 2(v&3) + {0,1}.  VGPRs 0-3 cover
// k = 8h..8h+7 and VGPRs 4-7 cover k = 16+8h..23+8h -> two 16B loads.
DEV v16bf frag_rowK(const unsigned short* p, int ld, int row0, int k0, int lane) {
  FragU f;
  const unsigned short* base =
      p + (size_t)(row0 + (lane & 15)) * ld + k0 + ((lane >> 4) << 3);
  f.q[0] = *(const uint4*)(base);
  f.q[1] = *(const uint4*)(base + 16);
  return f.v;
}

DEV v8f wmma_bf16(v16bf a, v16bf b, v8f c) {
  return __builtin_amdgcn_wmma_f32_16x16x32_bf16(
      /*neg_a=*/false, a, /*neg_b=*/false, b,
      /*c_mod=*/(short)0, c, /*reuse_a=*/false, /*reuse_b=*/false);
}

// ---------------------------------------------------------------------------
// Kernel 0: f32 -> bf16 conversion (x and weights into workspace)
// ---------------------------------------------------------------------------
__global__ void cvt_bf16_kernel(const float* __restrict__ s,
                                unsigned short* __restrict__ d, int n) {
  int i  = blockIdx.x * blockDim.x + threadIdx.x;
  int st = gridDim.x * blockDim.x;
  for (; i < n; i += st) d[i] = f2bf(s[i]);
}

// ---------------------------------------------------------------------------
// Kernel 1: fused QKV projections.  One workgroup = (batch b, 64-col slab n0).
// x slab is transposed once into LDS (Xt[n][c]) so every operand is rowK:
//   qb[b][n][r] = sum_c Xt[n][c] Wq[r][c]   (stored (N,R))
//   kb[b][n][r] = sum_c Xt[n][c] Wk[r][c]   (stored (N,R) -- transposed K!)
//   vb[b][o][n] = sum_c Wv[o][c] Xt[n][c]   (stored (C,N))
// 80 output tiles (Q:8, K:8, V:64) wave-strided over 16 waves; K-dim = 256.
// ---------------------------------------------------------------------------
__global__ __launch_bounds__(512) void qkv_kernel(
    const unsigned short* __restrict__ xb,
    const unsigned short* __restrict__ wq,
    const unsigned short* __restrict__ wk,
    const unsigned short* __restrict__ wv,
    unsigned short* __restrict__ qb,
    unsigned short* __restrict__ kb,
    unsigned short* __restrict__ vb) {
  __shared__ __align__(16) unsigned short Xt[64 * Cc];  // [n][c], 32KB

  int b = blockIdx.y, n0 = blockIdx.x * 64;
  int tid = threadIdx.x, lane = tid & 31, w = tid >> 5;
  int h = lane >> 4, nl = lane & 15;
  const unsigned short* xp = xb + (size_t)b * Cc * Nc;

  // transpose fill: coalesced 4B global reads, b16 LDS scatter (once per WG)
  for (int q = tid; q < 8192; q += 512) {
    int c = q >> 5, n2 = (q & 31) << 1;
    unsigned int val = *(const unsigned int*)(xp + (size_t)c * Nc + n0 + n2);
    Xt[n2 * Cc + c]       = (unsigned short)(val & 0xFFFFu);
    Xt[(n2 + 1) * Cc + c] = (unsigned short)(val >> 16);
  }
  __syncthreads();

  for (int t = w; t < 80; t += 16) {
    v8f acc = v8zero();
    if (t < 16) {                         // Q (t<8) / K (t>=8) output (64x32)
      const unsigned short* wgt = (t < 8) ? wq : wk;
      unsigned short*       dst = (t < 8) ? qb : kb;
      int tq = t & 7, tn = tq >> 1, tr = tq & 1;
#pragma unroll
      for (int kk = 0; kk < 8; ++kk)
        acc = wmma_bf16(frag_rowK(Xt, Cc, tn * 16, kk * 32, lane),
                        frag_rowK(wgt, Cc, tr * 16, kk * 32, lane), acc);
#pragma unroll
      for (int v = 0; v < 8; ++v) {
        int nr = n0 + tn * 16 + v + 8 * h;
        dst[((size_t)b * Nc + nr) * Rc + tr * 16 + nl] = f2bf(acc[v]);
      }
    } else {                                            // V output (256 x 64)
      int tv = t - 16, tc = tv >> 2, tn = tv & 3;
#pragma unroll
      for (int kk = 0; kk < 8; ++kk)
        acc = wmma_bf16(frag_rowK(wv, Cc, tc * 16, kk * 32, lane),
                        frag_rowK(Xt, Cc, tn * 16, kk * 32, lane), acc);
#pragma unroll
      for (int v = 0; v < 8; ++v) {
        int c = tc * 16 + v + 8 * h;
        vb[((size_t)b * Cc + c) * Nc + n0 + tn * 16 + nl] = f2bf(acc[v]);
      }
    }
  }
}

// ---------------------------------------------------------------------------
// Kernel 2: flash attention + epilogue.  One workgroup = (b, 64-query block).
// Streams 64 j-blocks; online softmax (8 threads/row); O accumulator
// (256 x 64 f32) in registers (wave w owns c-strip [16w,16w+16), 4 i-tiles).
// out = gamma * (V P^T / l) + x
// ---------------------------------------------------------------------------
__global__ __launch_bounds__(512) void attn_kernel(
    const unsigned short* __restrict__ qb,
    const unsigned short* __restrict__ kb,
    const unsigned short* __restrict__ vb,
    const float* __restrict__ x,
    const float* __restrict__ gamma,
    float* __restrict__ out) {
  __shared__ __align__(16) unsigned short Qs[64 * 32];   // [i][r]   4KB
  __shared__ __align__(16) unsigned short Kt[64 * 32];   // [j][r]   4KB
  __shared__ __align__(16) unsigned short Vs[256 * 64];  // [c][j]  32KB
  __shared__ __align__(16) unsigned short Ps[64 * 64];   // [i][j]   8KB
  __shared__ float Ss[64 * 64];                          // logits  16KB
  __shared__ float red[512];                             // reduce   2KB
  __shared__ float mS[64], lS[64], aS[64];

  int b = blockIdx.y, i0 = blockIdx.x * 64;
  int tid = threadIdx.x, lane = tid & 31, w = tid >> 5;
  int h = lane >> 4, nl = lane & 15;
  const unsigned short* qp = qb + (size_t)b * Nc * Rc;
  const unsigned short* kp = kb + (size_t)b * Nc * Rc;
  const unsigned short* vp = vb + (size_t)b * Cc * Nc;

  if (tid < 256) {                                       // Q block once
    int row = tid >> 2, off = (tid & 3) << 3;
    *(uint4*)(Qs + row * 32 + off) =
        *(const uint4*)(qp + (size_t)(i0 + row) * Rc + off);
  }
  if (tid < 64) { mS[tid] = -3.4e38f; lS[tid] = 0.f; }

  v8f acc[4];
#pragma unroll
  for (int t = 0; t < 4; ++t) acc[t] = v8zero();
  int cw = w * 16;
  int ir = tid >> 3, sub = tid & 7;                      // softmax mapping

  for (int jb = 0; jb < Nc / 64; ++jb) {
    int j0 = jb * 64;
    __syncthreads();                                     // prev O-GEMM done
    if (tid < 256) {                                     // K^T block 64x32
      int row = tid >> 2, off = (tid & 3) << 3;
      *(uint4*)(Kt + row * 32 + off) =
          *(const uint4*)(kp + (size_t)(j0 + row) * Rc + off);
    }
    for (int q = tid; q < 2048; q += 512) {              // V block 256x64
      int row = q >> 3, off = (q & 7) << 3;
      *(uint4*)(Vs + row * 64 + off) =
          *(const uint4*)(vp + (size_t)row * Nc + j0 + off);
    }
    if (jb + 1 < Nc / 64) {                              // global_prefetch_b8
      int jn = j0 + 64;
      if (tid < 256) __builtin_prefetch(vp + (size_t)tid * Nc + jn, 0, 1);
      if (tid < 64)  __builtin_prefetch(kp + (size_t)(jn + tid) * Rc, 0, 1);
    }
    __syncthreads();

    {  // S = Q K^T : 16 tiles, one per wave; K-dim = R = 32 -> single WMMA
      int ti = w >> 2, tj = w & 3;
      v8f s = wmma_bf16(frag_rowK(Qs, 32, ti * 16, 0, lane),
                        frag_rowK(Kt, 32, tj * 16, 0, lane), v8zero());
#pragma unroll
      for (int v = 0; v < 8; ++v)
        Ss[(ti * 16 + v + 8 * h) * 64 + tj * 16 + nl] = s[v];
    }
    __syncthreads();

    // ---- online softmax: 8 threads per row ----
    {
      const float* row = Ss + ir * 64 + sub * 8;
      float lm = -3.4e38f;
#pragma unroll
      for (int j = 0; j < 8; ++j) lm = fmaxf(lm, row[j]);
      red[tid] = lm;
      __syncthreads();
      float mb = red[ir * 8];
#pragma unroll
      for (int j = 1; j < 8; ++j) mb = fmaxf(mb, red[ir * 8 + j]);
      float mo = mS[ir];
      float mn = fmaxf(mo, mb);
      float s = 0.f;
      unsigned short* pr = Ps + ir * 64 + sub * 8;
#pragma unroll
      for (int j = 0; j < 8; ++j) {
        float e = __expf(row[j] - mn);
        s += e;
        pr[j] = f2bf(e);
      }
      __syncthreads();
      red[tid] = s;
      __syncthreads();
      if (sub == 0) {
        float sum = red[ir * 8];
#pragma unroll
        for (int j = 1; j < 8; ++j) sum += red[ir * 8 + j];
        float al = __expf(mo - mn);
        mS[ir] = mn;
        lS[ir] = lS[ir] * al + sum;
        aS[ir] = al;
      }
    }
    __syncthreads();

    // rescale accumulator (column i = lane -> per-lane scalar) then O += V P^T
#pragma unroll
    for (int t = 0; t < 4; ++t) {
      float al = aS[t * 16 + nl];
#pragma unroll
      for (int v = 0; v < 8; ++v) acc[t][v] *= al;
    }
#pragma unroll
    for (int kk = 0; kk < 2; ++kk) {
      v16bf a = frag_rowK(Vs, 64, cw, kk * 32, lane);
#pragma unroll
      for (int t = 0; t < 4; ++t)
        acc[t] = wmma_bf16(a, frag_rowK(Ps, 64, t * 16, kk * 32, lane), acc[t]);
    }
  }
  __syncthreads();

  float g = gamma[0];
#pragma unroll
  for (int t = 0; t < 4; ++t) {
    int il = t * 16 + nl;
    float inv = 1.f / lS[il];
    size_t col = (size_t)i0 + il;
#pragma unroll
    for (int v = 0; v < 8; ++v) {
      size_t idx = ((size_t)b * Cc + cw + v + 8 * h) * Nc + col;
      out[idx] = g * acc[t][v] * inv + x[idx];
    }
  }
}

// ---------------------------------------------------------------------------
extern "C" void kernel_launch(void* const* d_in, const int* in_sizes, int n_in,
                              void* d_out, int out_size, void* d_ws,
                              size_t ws_size, hipStream_t stream) {
  (void)in_sizes; (void)n_in; (void)out_size; (void)ws_size;
  const float* x     = (const float*)d_in[0];
  const float* Wq    = (const float*)d_in[1];
  const float* Wk    = (const float*)d_in[2];
  const float* Wv    = (const float*)d_in[3];
  const float* gamma = (const float*)d_in[4];
  float* out = (float*)d_out;

  char* ws = (char*)d_ws;
  size_t off = 0;
  unsigned short* xb  = (unsigned short*)(ws + off); off += (size_t)Bc * Cc * Nc * 2;
  unsigned short* qb  = (unsigned short*)(ws + off); off += (size_t)Bc * Nc * Rc * 2;
  unsigned short* kb  = (unsigned short*)(ws + off); off += (size_t)Bc * Nc * Rc * 2;
  unsigned short* vb  = (unsigned short*)(ws + off); off += (size_t)Bc * Cc * Nc * 2;
  unsigned short* wqb = (unsigned short*)(ws + off); off += (size_t)Rc * Cc * 2;
  unsigned short* wkb = (unsigned short*)(ws + off); off += (size_t)Rc * Cc * 2;
  unsigned short* wvb = (unsigned short*)(ws + off); off += (size_t)Cc * Cc * 2;

  cvt_bf16_kernel<<<4096, 256, 0, stream>>>(x,  xb,  Bc * Cc * Nc);
  cvt_bf16_kernel<<<32,   256, 0, stream>>>(Wq, wqb, Rc * Cc);
  cvt_bf16_kernel<<<32,   256, 0, stream>>>(Wk, wkb, Rc * Cc);
  cvt_bf16_kernel<<<256,  256, 0, stream>>>(Wv, wvb, Cc * Cc);

  qkv_kernel<<<dim3(Nc / 64, Bc), 512, 0, stream>>>(xb, wqb, wkb, wvb,
                                                    qb, kb, vb);
  attn_kernel<<<dim3(Nc / 64, Bc), 512, 0, stream>>>(qb, kb, vb, x, gamma, out);
}
